// HelixLinear_32615981646541
// MI455X (gfx1250) — matrix operand order repaired and checked by reading, not compiled, not run on hardware
//
#include <hip/hip_runtime.h>

// Problem constants (from reference setup_inputs)
#define M_DIM   8192              // B*S rows of x / out
#define K_DIM   4096              // IN
#define N_DIM   4096              // OUT
#define RANK    32
#define NNZ_CNT 65536

// GEMM tiling
#define KSTEP   64                // K per LDS stage
#define AROWS   128               // block tile M
#define BROWS   256               // block tile N
#define ASTR    72                // padded LDS row stride (elements)
#define LDS_A   (2 * AROWS * ASTR)   // elements, double buffered
#define LDS_B   (2 * BROWS * ASTR)
#define LDS_BYTES ((LDS_A + LDS_B) * 2)

typedef __attribute__((ext_vector_type(16))) __bf16 v16bf;
typedef __attribute__((ext_vector_type(8)))  __bf16 v8bf;
typedef __attribute__((ext_vector_type(8)))  float  v8f;
typedef __attribute__((ext_vector_type(4)))  int    v4i;

#if defined(__gfx1250__) && __has_builtin(__builtin_amdgcn_global_load_async_to_lds_b128)
#define HAVE_ASYNC 1
typedef v4i __attribute__((address_space(1)))* g128_ptr;
typedef v4i __attribute__((address_space(3)))* l128_ptr;
#else
#define HAVE_ASYNC 0
#endif

// 16-byte global -> LDS copy (async DMA on gfx1250, sync fallback otherwise)
__device__ __forceinline__ void cp16_g2l(const __bf16* g, __bf16* l) {
#if HAVE_ASYNC
  __builtin_amdgcn_global_load_async_to_lds_b128(
      (g128_ptr)g, (l128_ptr)l, 0, 0);
#else
  *(v8bf*)l = *(const v8bf*)g;
#endif
}

template <int N>
__device__ __forceinline__ void wait_async() {
#if HAVE_ASYNC
#if __has_builtin(__builtin_amdgcn_s_wait_asynccnt)
  __builtin_amdgcn_s_wait_asynccnt(N);
#else
  asm volatile("s_wait_asynccnt %0" ::"i"(N));
#endif
#endif
}

// ---------------------------------------------------------------------------
// Kernel 1: W32[o,i] = codebook[indices[o,i]] + sum_r (U[o,r]*s[r]) * Vt[r,i]
// ---------------------------------------------------------------------------
__global__ __launch_bounds__(256) void dequant_kernel(
    const float* __restrict__ codebook,
    const int*   __restrict__ indices,
    const float* __restrict__ U,
    const float* __restrict__ s,
    const float* __restrict__ Vt,
    float*       __restrict__ W32) {
  const int gid = blockIdx.x * 256 + threadIdx.x;
  const int o   = gid >> 12;            // / 4096 (uniform per block)
  const int i   = gid & (K_DIM - 1);
  float w = codebook[indices[gid]];
  const float* Ur = U + o * RANK;
#pragma unroll
  for (int r = 0; r < RANK; ++r) {
    w = fmaf(Ur[r] * s[r], Vt[r * K_DIM + i], w);
  }
  W32[gid] = w;
}

// ---------------------------------------------------------------------------
// Kernel 2: sparse sidecar scatter-add (duplicates possible -> atomics)
// ---------------------------------------------------------------------------
__global__ __launch_bounds__(256) void scatter_kernel(
    float*       __restrict__ W32,
    const int*   __restrict__ rows,
    const int*   __restrict__ cols,
    const float* __restrict__ deltas) {
  const int t = blockIdx.x * 256 + threadIdx.x;
  atomicAdd(W32 + (size_t)rows[t] * K_DIM + cols[t], deltas[t]);
}

// ---------------------------------------------------------------------------
// Kernel 3: fp32 -> bf16 pack, 8 elements/thread (b128 in, b128 out)
// ---------------------------------------------------------------------------
__global__ __launch_bounds__(256) void cvt_bf16_kernel(
    const float* __restrict__ src,
    __bf16*      __restrict__ dst) {
  const int t = blockIdx.x * 256 + threadIdx.x;
  const float4* p = (const float4*)src + (size_t)t * 2;
  const float4 f0 = p[0];
  const float4 f1 = p[1];
  v8bf o;
  o[0] = (__bf16)f0.x; o[1] = (__bf16)f0.y;
  o[2] = (__bf16)f0.z; o[3] = (__bf16)f0.w;
  o[4] = (__bf16)f1.x; o[5] = (__bf16)f1.y;
  o[6] = (__bf16)f1.z; o[7] = (__bf16)f1.w;
  *((v8bf*)dst + t) = o;
}

// ---------------------------------------------------------------------------
// Stage one KSTEP tile of A (128 x 64) and B (256 x 64) into LDS.
// 4 + 8 = 12 async b128 instructions per wave -> ASYNCcnt += 12.
// ---------------------------------------------------------------------------
__device__ __forceinline__ void stage_tile(const __bf16* __restrict__ Ag,
                                           const __bf16* __restrict__ Bg,
                                           __bf16* As, __bf16* Bs,
                                           int tid, int k0) {
#pragma unroll
  for (int c = 0; c < 4; ++c) {            // 1024 A chunks of 8 elems
    const int chunk = c * 256 + tid;
    const int row = chunk >> 3;            // 0..127
    const int ko  = (chunk & 7) * 8;       // 0..56
    cp16_g2l(Ag + (size_t)row * K_DIM + k0 + ko, As + row * ASTR + ko);
  }
#pragma unroll
  for (int c = 0; c < 8; ++c) {            // 2048 B chunks of 8 elems
    const int chunk = c * 256 + tid;
    const int row = chunk >> 3;            // 0..255
    const int ko  = (chunk & 7) * 8;
    cp16_g2l(Bg + (size_t)row * K_DIM + k0 + ko, Bs + row * ASTR + ko);
  }
}

// ---------------------------------------------------------------------------
// Kernel 4: WMMA bf16 GEMM with double-buffered async LDS staging.
//   out[m,n] = sum_k A[m,k] * W[n,k] + bias[n]
// Block = 256 threads = 8 waves (2M x 4N); block tile 128x256; wave 64x64.
// ---------------------------------------------------------------------------
__global__ __launch_bounds__(256) void wmma_gemm_kernel(
    const __bf16* __restrict__ A,
    const __bf16* __restrict__ W,
    const float*  __restrict__ bias,
    float*        __restrict__ C) {
  extern __shared__ __bf16 smem[];
  __bf16* As = smem;            // [2][AROWS][ASTR]
  __bf16* Bs = smem + LDS_A;    // [2][BROWS][ASTR]

  const int tid  = threadIdx.x;
  const int lane = tid & 31;
  const int wave = tid >> 5;
  const int h = lane >> 4;      // half-wave select
  const int r = lane & 15;      // row/col within fragment
  const int wn = (wave & 3) * 64;
  const int wm = (wave >> 2) * 64;
  const int mBlock = blockIdx.y * AROWS;
  const int nBlock = blockIdx.x * BROWS;

  const __bf16* Ag = A + (size_t)mBlock * K_DIM;
  const __bf16* Bg = W + (size_t)nBlock * K_DIM;

  v8f acc[4][4];
#pragma unroll
  for (int i = 0; i < 4; ++i)
#pragma unroll
    for (int j = 0; j < 4; ++j) acc[i][j] = v8f{};

  // Prologue: stage tile 0 into buffer 0.
  stage_tile(Ag, Bg, As, Bs, tid, 0);

  const int NT = K_DIM / KSTEP;  // 64
  for (int kt = 0; kt < NT; ++kt) {
    const int cur = kt & 1;
    if (kt + 1 < NT) {
      // Overlap: issue next tile's DMA, then wait only for the older 12.
      stage_tile(Ag, Bg, As + (cur ^ 1) * AROWS * ASTR,
                 Bs + (cur ^ 1) * BROWS * ASTR, tid, (kt + 1) * KSTEP);
      wait_async<12>();
    } else {
      wait_async<0>();
    }
    __syncthreads();

    const __bf16* Asb = As + cur * AROWS * ASTR;
    const __bf16* Bsb = Bs + cur * BROWS * ASTR;
#pragma unroll
    for (int kk = 0; kk < KSTEP; kk += 32) {
      v16bf a[4], b[4];
#pragma unroll
      for (int i = 0; i < 4; ++i) {
        // A lane(h,r): row wm+i*16+r, K elems {kk+h*8..} u {kk+16+h*8..}
        const __bf16* p = Asb + (wm + i * 16 + r) * ASTR + kk + h * 8;
        v8bf lo = *(const v8bf*)p;
        v8bf hi = *(const v8bf*)(p + 16);
        a[i] = __builtin_shufflevector(lo, hi, 0, 1, 2, 3, 4, 5, 6, 7,
                                               8, 9, 10, 11, 12, 13, 14, 15);
      }
#pragma unroll
      for (int j = 0; j < 4; ++j) {
        // B lane(h,r): col wn+j*16+r = W row, K elems {kk+h*8..} u {kk+16+h*8..}
        const __bf16* p = Bsb + (wn + j * 16 + r) * ASTR + kk + h * 8;
        v8bf lo = *(const v8bf*)p;
        v8bf hi = *(const v8bf*)(p + 16);
        b[j] = __builtin_shufflevector(lo, hi, 0, 1, 2, 3, 4, 5, 6, 7,
                                               8, 9, 10, 11, 12, 13, 14, 15);
      }
#pragma unroll
      for (int i = 0; i < 4; ++i)
#pragma unroll
        for (int j = 0; j < 4; ++j)
          acc[i][j] = __builtin_amdgcn_wmma_f32_16x16x32_bf16(
              false, a[i], false, b[j], (short)0, acc[i][j], false, false);
    }
    __syncthreads();
  }

  // Epilogue: lane(h,r) of fragment (i,j) owns n = nBlock+wn+j*16+r,
  // m = mBlock+wm+i*16+h*8+v for v in 0..7.
#pragma unroll
  for (int j = 0; j < 4; ++j) {
    const int n = nBlock + wn + j * 16 + r;
    const float bj = bias[n];
#pragma unroll
    for (int i = 0; i < 4; ++i) {
      const int m0 = mBlock + wm + i * 16 + h * 8;
      float* po = C + (size_t)m0 * N_DIM + n;
#pragma unroll
      for (int v = 0; v < 8; ++v)
        po[(size_t)v * N_DIM] = acc[i][j][v] + bj;
    }
  }
}

// ---------------------------------------------------------------------------
extern "C" void kernel_launch(void* const* d_in, const int* in_sizes, int n_in,
                              void* d_out, int out_size, void* d_ws, size_t ws_size,
                              hipStream_t stream) {
  const float* x        = (const float*)d_in[0];
  const float* codebook = (const float*)d_in[1];
  const int*   indices  = (const int*)d_in[2];
  const int*   srows    = (const int*)d_in[3];
  const int*   scols    = (const int*)d_in[4];
  const float* sdeltas  = (const float*)d_in[5];
  const float* svd_U    = (const float*)d_in[6];
  const float* svd_s    = (const float*)d_in[7];
  const float* svd_Vt   = (const float*)d_in[8];
  const float* bias     = (const float*)d_in[9];
  float* out = (float*)d_out;

  // Workspace layout: W32 (fp32 N*K) | W bf16 (N*K) | x bf16 (M*K)
  char* ws = (char*)d_ws;
  float*  W32 = (float*)ws;                                   // 64 MB
  __bf16* Wb  = (__bf16*)(ws + (size_t)N_DIM * K_DIM * 4);    // 32 MB
  __bf16* Xb  = (__bf16*)(ws + (size_t)N_DIM * K_DIM * 6);    // 64 MB

  dequant_kernel<<<(N_DIM * K_DIM) / 256, 256, 0, stream>>>(
      codebook, indices, svd_U, svd_s, svd_Vt, W32);

  scatter_kernel<<<NNZ_CNT / 256, 256, 0, stream>>>(W32, srows, scols, sdeltas);

  cvt_bf16_kernel<<<((size_t)N_DIM * K_DIM / 8) / 256, 256, 0, stream>>>(W32, Wb);
  cvt_bf16_kernel<<<((size_t)M_DIM * K_DIM / 8) / 256, 256, 0, stream>>>(x, Xb);

  dim3 grid(N_DIM / BROWS, M_DIM / AROWS);
  wmma_gemm_kernel<<<grid, 256, LDS_BYTES, stream>>>(Xb, Wb, bias, out);
}
